// get_model_5257039970997
// MI455X (gfx1250) — matrix operand order repaired and checked by reading, not compiled, not run on hardware
//
#include <hip/hip_runtime.h>
#include <hip/hip_bf16.h>
#include <hip/hip_fp16.h>
#include <math.h>

// ---------------------------------------------------------------------------
// PointNet++ (MSG) forward for MI455X / gfx1250, wave32 + WMMA f16->f32.
// GEMMs: 8 waves/block, each wave = 1 M-tile x NB N-tiles (register blocking),
// weight slice staged in LDS (async-to-LDS when toolchain exposes it).
// ---------------------------------------------------------------------------

#define BSZ 32
#define N1  8192
#define NP1 128
#define NP2 64
#define BN_SCALE_C 0.9999950000374997f

typedef __attribute__((ext_vector_type(16))) _Float16 v16h;
typedef __attribute__((ext_vector_type(8)))  _Float16 v8h;
typedef __attribute__((ext_vector_type(8)))  float    v8f;

#if defined(__has_builtin)
#  if __has_builtin(__builtin_amdgcn_global_load_async_to_lds_b128) && \
      __has_builtin(__builtin_amdgcn_s_wait_asynccnt)
#    define HAVE_ASYNC_LDS 1
#  endif
#endif
#ifndef HAVE_ASYNC_LDS
#  define HAVE_ASYNC_LDS 0
#endif

#if HAVE_ASYNC_LDS
// Builtin parameter types (from hipcc diagnostic): AS1 / AS3 pointers to a
// 16-byte int vector.
typedef int v4i_g __attribute__((vector_size(16)));
typedef __attribute__((address_space(1))) v4i_g as1_v4i;
typedef __attribute__((address_space(3))) v4i_g as3_v4i;
#endif

// 16B global->LDS copy (per lane). Async (ASYNCcnt-tracked) when available.
__device__ __forceinline__ void cp16_g2l(const _Float16* __restrict__ g,
                                         _Float16* __restrict__ l) {
#if HAVE_ASYNC_LDS
  __builtin_amdgcn_global_load_async_to_lds_b128((as1_v4i*)g, (as3_v4i*)l, 0, 0);
#else
  *(v8h*)l = *(const v8h*)g;
#endif
}
__device__ __forceinline__ void cp_g2l_fence() {
#if HAVE_ASYNC_LDS
  __builtin_amdgcn_s_wait_asynccnt(0);
#endif
  __syncthreads();
}

__device__ __forceinline__ v16h cat16(v8h lo, v8h hi) {
  return __builtin_shufflevector(lo, hi, 0, 1, 2, 3, 4, 5, 6, 7,
                                 8, 9, 10, 11, 12, 13, 14, 15);
}

// ---------------------------------------------------------------------------
// Small utility kernels
// ---------------------------------------------------------------------------
__global__ void k_split(const float* __restrict__ xyz, float* __restrict__ coords,
                        _Float16* __restrict__ normh) {
  int i = blockIdx.x * blockDim.x + threadIdx.x;       // over BSZ*N1
  if (i >= BSZ * N1) return;
  int b = i / N1, n = i - b * N1;
  const float* src = xyz + (size_t)b * 6 * N1 + n;
  float* c = coords + (size_t)i * 3;
  _Float16* h = normh + (size_t)i * 3;
  c[0] = src[0];
  c[1] = src[(size_t)N1];
  c[2] = src[(size_t)2 * N1];
  h[0] = (_Float16)src[(size_t)3 * N1];
  h[1] = (_Float16)src[(size_t)4 * N1];
  h[2] = (_Float16)src[(size_t)5 * N1];
}

__global__ void k_zero_f32(float* __restrict__ p, int n) {
  int i = blockIdx.x * blockDim.x + threadIdx.x;
  if (i < n) p[i] = 0.0f;
}

__global__ void k_f32_to_f16(const float* __restrict__ s, _Float16* __restrict__ d, int n) {
  int i = blockIdx.x * blockDim.x + threadIdx.x;
  if (i < n) d[i] = (_Float16)s[i];
}

// W (cout x cin) f32 -> Wh (cout x cinPad) f16, zero padded
__global__ void k_cvt_w(const float* __restrict__ W, _Float16* __restrict__ Wh,
                        int cin, int cinPad, int n) {
  int i = blockIdx.x * blockDim.x + threadIdx.x;
  if (i >= n) return;
  int o = i / cinPad, c = i - o * cinPad;
  Wh[i] = (c < cin) ? (_Float16)W[(size_t)o * cin + c] : (_Float16)0.0f;
}

// sa3 input rows: [new_xyz2(3) | pool2(640) | zero-pad to 672]
__global__ void k_build_sa3in(const float* __restrict__ nx2, const float* __restrict__ pool2,
                              _Float16* __restrict__ out) {
  int i = blockIdx.x * blockDim.x + threadIdx.x;
  const int CIN = 672;
  if (i >= BSZ * NP2 * CIN) return;
  int row = i / CIN, c = i - row * CIN;
  float v = 0.0f;
  if (c < 3)        v = nx2[(size_t)row * 3 + c];
  else if (c < 643) v = pool2[(size_t)row * 640 + (c - 3)];
  out[i] = (_Float16)v;
}

__global__ void k_gather_xyz(const float* __restrict__ pts, const int* __restrict__ idx,
                             int N, int np, float* __restrict__ out) {
  int i = blockIdx.x * blockDim.x + threadIdx.x;
  if (i >= BSZ * np) return;
  int b = i / np;
  int j = idx[i];
  const float* s = pts + ((size_t)b * N + j) * 3;
  out[(size_t)i * 3 + 0] = s[0];
  out[(size_t)i * 3 + 1] = s[1];
  out[(size_t)i * 3 + 2] = s[2];
}

// ---------------------------------------------------------------------------
// Farthest point sampling: one workgroup per batch, dist[] in LDS.
// Matches reference: fps[0]=0, record-then-update, argmax first-index tiebreak.
// ---------------------------------------------------------------------------
__global__ void k_fps(const float* __restrict__ coords, int N, int npoint,
                      int* __restrict__ fpsIdx) {
  __shared__ float dist[8192];
  __shared__ float rv[1024];
  __shared__ int   ri[1024];
  __shared__ int   far_;
  int b = blockIdx.x, t = threadIdx.x, T = blockDim.x;
  const float* C = coords + (size_t)b * N * 3;
  for (int i = t; i < N; i += T) dist[i] = 1e10f;
  if (t == 0) far_ = 0;
  __syncthreads();
  for (int s = 0; s < npoint; ++s) {
    int f = far_;
    if (t == 0) fpsIdx[b * npoint + s] = f;
    float cx = C[(size_t)f * 3], cy = C[(size_t)f * 3 + 1], cz = C[(size_t)f * 3 + 2];
    float bv = -1.0f; int bi = 0x7fffffff;
    for (int i = t; i < N; i += T) {
      float dx = C[(size_t)i * 3] - cx;
      float dy = C[(size_t)i * 3 + 1] - cy;
      float dz = C[(size_t)i * 3 + 2] - cz;
      float d = dx * dx + dy * dy + dz * dz;
      float nd = fminf(dist[i], d);
      dist[i] = nd;
      if (nd > bv) { bv = nd; bi = i; }
    }
    rv[t] = bv; ri[t] = bi;
    __syncthreads();
    for (int o = T >> 1; o > 0; o >>= 1) {
      if (t < o) {
        if (rv[t + o] > rv[t] || (rv[t + o] == rv[t] && ri[t + o] < ri[t])) {
          rv[t] = rv[t + o]; ri[t] = ri[t + o];
        }
      }
      __syncthreads();
    }
    if (t == 0) far_ = ri[0];
    __syncthreads();
  }
}

// ---------------------------------------------------------------------------
// Ball query: one thread per (b,s). First K indices with d^2<=r^2, pad w/first.
// ---------------------------------------------------------------------------
__global__ void k_query_ball(const float* __restrict__ coords, const float* __restrict__ nxyz,
                             int N, int np, int K, float r2, int* __restrict__ gi) {
  int t = blockIdx.x * blockDim.x + threadIdx.x;
  if (t >= BSZ * np) return;
  int b = t / np;
  const float* C = coords + (size_t)b * N * 3;
  const float* q = nxyz + (size_t)t * 3;
  int* out = gi + (size_t)t * K;
  float qx = q[0], qy = q[1], qz = q[2];
  int cnt = 0, first = -1;
  for (int j = 0; j < N && cnt < K; ++j) {
    float dx = C[(size_t)j * 3] - qx;
    float dy = C[(size_t)j * 3 + 1] - qy;
    float dz = C[(size_t)j * 3 + 2] - qz;
    if (dx * dx + dy * dy + dz * dz <= r2) {
      if (first < 0) first = j;
      out[cnt++] = j;
    }
  }
  if (first < 0) first = 0;
  for (int i = cnt; i < K; ++i) out[i] = first;
}

// ---------------------------------------------------------------------------
// Dense GEMM, register-blocked: block = 8 waves = 8 M-tiles x NB N-tiles.
// W slice for the block's NB N-tiles is staged in LDS (async when available);
// each wave loads its A fragment once per k-step and reuses it for NB WMMAs.
//   D = relu((A*W^T + bias) * (gamma*BN_SCALE) + beta)
// poolK==0 : store f16 rows->outH (stride cout)
// poolK>0  : fused maxpool over groups of poolK rows via uint atomicMax
//            (post-ReLU values >= 0) into outF[row/poolK * poolStride + chanOff].
// rows % 128 == 0 ; cin % 32 == 0 ; (cout/16) % NB == 0  (no tails, EXEC all-1)
// ---------------------------------------------------------------------------
template <int NB>
__global__ void k_gemm_wmma_t(const _Float16* __restrict__ A, const _Float16* __restrict__ W,
                              const float* __restrict__ bias, const float* __restrict__ gamma,
                              const float* __restrict__ beta,
                              int rows, int cin, int cout, int poolK,
                              _Float16* __restrict__ outH, float* __restrict__ outF,
                              int poolStride, int chanOff) {
  extern __shared__ __align__(16) _Float16 ldsW[];   // [NB*16][cin]
  int ntN = cout >> 4;
  int nGroups = ntN / NB;
  int mBlk = blockIdx.x / nGroups;
  int tN0 = (blockIdx.x - mBlk * nGroups) * NB;

  // cooperative W slice copy: rows [tN0*16, tN0*16+NB*16) x cin, contiguous
  {
    const _Float16* src = W + (size_t)(tN0 * 16) * cin;
    int halfs = NB * 16 * cin;
    for (int e = threadIdx.x * 8; e < halfs; e += blockDim.x * 8)
      cp16_g2l(src + e, ldsW + e);
  }
  cp_g2l_fence();

  int wv = threadIdx.x >> 5, lane = threadIdx.x & 31;
  int tM = mBlk * 8 + wv;
  int m = lane & 15;
  const _Float16* Arow = A + (size_t)(tM * 16 + m) * cin + ((lane < 16) ? 0 : 8);
  const _Float16* Bbase = ldsW + (size_t)m * cin + ((lane < 16) ? 0 : 8);

  v8f acc[NB];
#pragma unroll
  for (int j = 0; j < NB; ++j) acc[j] = v8f{};

  for (int kc = 0; kc < cin; kc += 32) {
    v16h a = cat16(*(const v8h*)(Arow + kc), *(const v8h*)(Arow + kc + 16));
#pragma unroll
    for (int j = 0; j < NB; ++j) {
      const _Float16* br = Bbase + (size_t)(j * 16) * cin + kc;     // ds_load_b128 x2
      v16h bv = cat16(*(const v8h*)br, *(const v8h*)(br + 16));
      acc[j] = __builtin_amdgcn_wmma_f32_16x16x32_f16(false, a, false, bv,
                                                      (short)0, acc[j], false, false);
    }
  }

  int mbase = tM * 16 + ((lane < 16) ? 0 : 8);
#pragma unroll
  for (int j = 0; j < NB; ++j) {
    int col = (tN0 + j) * 16 + m;
    float sc = gamma[col] * BN_SCALE_C;
    float bo = bias[col], bt = beta[col];
#pragma unroll
    for (int r = 0; r < 8; ++r) {
      int grow = mbase + r;
      float y = (acc[j][r] + bo) * sc + bt;
      y = y > 0.0f ? y : 0.0f;
      if (poolK == 0) {
        outH[(size_t)grow * cout + col] = (_Float16)y;
      } else {
        int grp = grow / poolK;
        unsigned int* p = (unsigned int*)(outF + (size_t)grp * poolStride + chanOff + col);
        atomicMax(p, __float_as_uint(y));
      }
    }
  }
}

// ---------------------------------------------------------------------------
// Layer-1 GEMM with fused neighbor gather (A row (b,s,k) =
// [ pts[b, gi[b,s,k], 0..pc) | coords[gi]-new_xyz[b,s] (3) | zero pad ]),
// same LDS-staged W + NB register blocking; the expensive gathered-A build
// is amortized over NB WMMAs.
// ---------------------------------------------------------------------------
template <int NB>
__global__ void k_gemm_gather_t(const _Float16* __restrict__ pts, int pc,
                                const float* __restrict__ coords,
                                const float* __restrict__ nxyz,
                                const int* __restrict__ gi,
                                int Npts, int np, int K,
                                const _Float16* __restrict__ W,
                                const float* __restrict__ bias,
                                const float* __restrict__ gamma,
                                const float* __restrict__ beta,
                                int rows, int cinPad, int cout,
                                _Float16* __restrict__ outH) {
  extern __shared__ __align__(16) _Float16 ldsW[];   // [NB*16][cinPad]
  int ntN = cout >> 4;
  int nGroups = ntN / NB;
  int mBlk = blockIdx.x / nGroups;
  int tN0 = (blockIdx.x - mBlk * nGroups) * NB;

  {
    const _Float16* src = W + (size_t)(tN0 * 16) * cinPad;
    int halfs = NB * 16 * cinPad;
    for (int e = threadIdx.x * 8; e < halfs; e += blockDim.x * 8)
      cp16_g2l(src + e, ldsW + e);
  }
  cp_g2l_fence();

  int wv = threadIdx.x >> 5, lane = threadIdx.x & 31;
  int tM = mBlk * 8 + wv;
  int m = lane & 15;
  int row = tM * 16 + m;
  int npk = np * K;
  int b = row / npk;
  int r2 = row - b * npk;
  int s = r2 / K;
  int idx = gi[(size_t)(b * np + s) * K + (r2 - s * K)];
  const _Float16* prow = pts + ((size_t)b * Npts + idx) * pc;
  const float* cr = coords + ((size_t)b * Npts + idx) * 3;
  const float* q = nxyz + (size_t)(b * np + s) * 3;
  _Float16 dxh[3] = { (_Float16)(cr[0] - q[0]), (_Float16)(cr[1] - q[1]),
                      (_Float16)(cr[2] - q[2]) };
  int klo = (lane < 16) ? 0 : 8;
  const _Float16* Bbase = ldsW + (size_t)m * cinPad + ((lane < 16) ? 0 : 8);

  v8f acc[NB];
#pragma unroll
  for (int j = 0; j < NB; ++j) acc[j] = v8f{};

  for (int kc = 0; kc < cinPad; kc += 32) {
    v16h a;
#pragma unroll
    for (int e = 0; e < 16; ++e) {
      int k = kc + klo + ((e < 8) ? e : e + 8);
      _Float16 v = (_Float16)0.0f;
      if (k < pc)          v = prow[k];
      else if (k < pc + 3) v = dxh[k - pc];
      a[e] = v;
    }
#pragma unroll
    for (int j = 0; j < NB; ++j) {
      const _Float16* br = Bbase + (size_t)(j * 16) * cinPad + kc;
      v16h bv = cat16(*(const v8h*)br, *(const v8h*)(br + 16));
      acc[j] = __builtin_amdgcn_wmma_f32_16x16x32_f16(false, a, false, bv,
                                                      (short)0, acc[j], false, false);
    }
  }

  int mbase = tM * 16 + ((lane < 16) ? 0 : 8);
#pragma unroll
  for (int j = 0; j < NB; ++j) {
    int col = (tN0 + j) * 16 + m;
    float sc = gamma[col] * BN_SCALE_C;
    float bo = bias[col], bt = beta[col];
#pragma unroll
    for (int r = 0; r < 8; ++r) {
      float y = (acc[j][r] + bo) * sc + bt;
      y = y > 0.0f ? y : 0.0f;
      outH[(size_t)(mbase + r) * cout + col] = (_Float16)y;
    }
  }
}

// ---------------------------------------------------------------------------
// FC head: 1024->512->256->16 + log_softmax, one workgroup per batch row.
// ---------------------------------------------------------------------------
__global__ void k_fc_head(const float* __restrict__ l3,
                          const float* __restrict__ w1, const float* __restrict__ b1,
                          const float* __restrict__ g1, const float* __restrict__ be1,
                          const float* __restrict__ w2, const float* __restrict__ b2,
                          const float* __restrict__ g2, const float* __restrict__ be2,
                          const float* __restrict__ w3, const float* __restrict__ b3,
                          float* __restrict__ out) {
  __shared__ float x[1024];
  __shared__ float h1[512];
  __shared__ float h2[256];
  __shared__ float lg[16];
  int b = blockIdx.x, t = threadIdx.x;
  for (int i = t; i < 1024; i += 256) x[i] = l3[(size_t)b * 1024 + i];
  __syncthreads();
  for (int o = t; o < 512; o += 256) {
    float a = 0.0f;
    const float* wr = w1 + (size_t)o * 1024;
    for (int k = 0; k < 1024; ++k) a += x[k] * wr[k];
    a = (a + b1[o]) * (g1[o] * BN_SCALE_C) + be1[o];
    h1[o] = fmaxf(a, 0.0f);
  }
  __syncthreads();
  for (int o = t; o < 256; o += 256) {
    float a = 0.0f;
    const float* wr = w2 + (size_t)o * 512;
    for (int k = 0; k < 512; ++k) a += h1[k] * wr[k];
    a = (a + b2[o]) * (g2[o] * BN_SCALE_C) + be2[o];
    h2[o] = fmaxf(a, 0.0f);
  }
  __syncthreads();
  if (t < 16) {
    float a = 0.0f;
    const float* wr = w3 + (size_t)t * 256;
    for (int k = 0; k < 256; ++k) a += h2[k] * wr[k];
    lg[t] = a + b3[t];
  }
  __syncthreads();
  if (t == 0) {
    float mx = lg[0];
    for (int i = 1; i < 16; ++i) mx = fmaxf(mx, lg[i]);
    float sum = 0.0f;
    for (int i = 0; i < 16; ++i) sum += expf(lg[i] - mx);
    float lse = mx + logf(sum);
    for (int i = 0; i < 16; ++i) out[(size_t)b * 16 + i] = lg[i] - lse;
  }
}

// ---------------------------------------------------------------------------
// Host orchestration
// ---------------------------------------------------------------------------
extern "C" void kernel_launch(void* const* d_in, const int* in_sizes, int n_in,
                              void* d_out, int out_size, void* d_ws, size_t ws_size,
                              hipStream_t stream) {
  (void)in_sizes; (void)n_in; (void)out_size; (void)ws_size;
  const float* xyz = (const float*)d_in[0];

  struct Lay { const float *w, *b, *g, *be; int cin, cout, cinPad; };
  Lay L[21];
  int ii = 1, li = 0;
  auto addChain = [&](int cin, const int* outs, int n) {
    for (int j = 0; j < n; ++j) {
      L[li].w  = (const float*)d_in[ii + 0];
      L[li].b  = (const float*)d_in[ii + 1];
      L[li].g  = (const float*)d_in[ii + 2];
      L[li].be = (const float*)d_in[ii + 3];
      ii += 4;
      L[li].cin = cin; L[li].cout = outs[j];
      L[li].cinPad = ((cin + 31) / 32) * 32;
      cin = outs[j];
      ++li;
    }
  };
  { int o[3] = {32, 32, 64};     addChain(6, o, 3); }    // sa1 b0
  { int o[3] = {64, 64, 128};    addChain(6, o, 3); }    // sa1 b1
  { int o[3] = {64, 96, 128};    addChain(6, o, 3); }    // sa1 b2
  { int o[3] = {64, 64, 128};    addChain(323, o, 3); }  // sa2 b0
  { int o[3] = {128, 128, 256};  addChain(323, o, 3); }  // sa2 b1
  { int o[3] = {128, 128, 256};  addChain(323, o, 3); }  // sa2 b2
  { int o[3] = {256, 512, 1024}; addChain(643, o, 3); }  // sa3
  const float* fc1w = (const float*)d_in[ii++]; const float* fc1b = (const float*)d_in[ii++];
  const float* bn1g = (const float*)d_in[ii++]; const float* bn1b = (const float*)d_in[ii++];
  const float* fc2w = (const float*)d_in[ii++]; const float* fc2b = (const float*)d_in[ii++];
  const float* bn2g = (const float*)d_in[ii++]; const float* bn2b = (const float*)d_in[ii++];
  const float* fc3w = (const float*)d_in[ii++]; const float* fc3b = (const float*)d_in[ii++];

  // ---- workspace carve-out (~150 MB) ----
  char* wsp = (char*)d_ws;
  size_t off = 0;
  auto walloc = [&](size_t bytes) -> void* {
    void* p = wsp + off;
    off += (bytes + 255) & ~(size_t)255;
    return p;
  };
  float*    coords = (float*)   walloc((size_t)BSZ * N1 * 3 * 4);
  _Float16* normh  = (_Float16*)walloc((size_t)BSZ * N1 * 3 * 2);
  int*      fps1   = (int*)     walloc((size_t)BSZ * NP1 * 4);
  float*    nx1    = (float*)   walloc((size_t)BSZ * NP1 * 3 * 4);
  int*      fps2   = (int*)     walloc((size_t)BSZ * NP2 * 4);
  float*    nx2    = (float*)   walloc((size_t)BSZ * NP2 * 3 * 4);
  int*      gi     = (int*)     walloc((size_t)262144 * 4);
  _Float16* act0   = (_Float16*)walloc((size_t)262144 * 128 * 2);
  _Float16* act1   = (_Float16*)walloc((size_t)262144 * 128 * 2);
  float*    pool1  = (float*)   walloc((size_t)BSZ * NP1 * 320 * 4);
  _Float16* pts2   = (_Float16*)walloc((size_t)BSZ * NP1 * 320 * 2);
  float*    pool2  = (float*)   walloc((size_t)BSZ * NP2 * 640 * 4);
  _Float16* sa3in  = (_Float16*)walloc((size_t)BSZ * NP2 * 672 * 2);
  _Float16* wh[21];
  for (int j = 0; j < 21; ++j)
    wh[j] = (_Float16*)walloc((size_t)L[j].cout * L[j].cinPad * 2);

  auto cdiv = [](int a, int b) { return (a + b - 1) / b; };

  // ---- weights f32 -> f16 padded ----
  for (int j = 0; j < 21; ++j) {
    int n = L[j].cout * L[j].cinPad;
    k_cvt_w<<<cdiv(n, 256), 256, 0, stream>>>(L[j].w, wh[j], L[j].cin, L[j].cinPad, n);
  }

  // ---- geometry pipeline ----
  k_split<<<cdiv(BSZ * N1, 256), 256, 0, stream>>>(xyz, coords, normh);
  k_fps<<<BSZ, 1024, 0, stream>>>(coords, N1, NP1, fps1);
  k_gather_xyz<<<cdiv(BSZ * NP1, 256), 256, 0, stream>>>(coords, fps1, N1, NP1, nx1);
  k_fps<<<BSZ, 128, 0, stream>>>(nx1, NP1, NP2, fps2);
  k_gather_xyz<<<cdiv(BSZ * NP2, 256), 256, 0, stream>>>(nx1, fps2, NP1, NP2, nx2);

  // NB = N-tiles per wave; all ntN are even, NB=4 when divisible and LDS<=64KB
  auto pickNB = [](int ntN, int cin) {
    return (ntN % 4 == 0 && 128 * cin <= 65536) ? 4 : 2;
  };

  auto gemm = [&](const _Float16* A, int lidx, int rows, int cin, int poolK,
                  _Float16* oh, float* of, int ps, int co) {
    int cout = L[lidx].cout, ntN = cout >> 4;
    int NB = pickNB(ntN, cin);
    int blocks = (rows / 128) * (ntN / NB);
    size_t shmem = (size_t)NB * 16 * cin * 2;
    if (NB == 4)
      k_gemm_wmma_t<4><<<blocks, 256, shmem, stream>>>(A, wh[lidx], L[lidx].b, L[lidx].g,
          L[lidx].be, rows, cin, cout, poolK, oh, of, ps, co);
    else
      k_gemm_wmma_t<2><<<blocks, 256, shmem, stream>>>(A, wh[lidx], L[lidx].b, L[lidx].g,
          L[lidx].be, rows, cin, cout, poolK, oh, of, ps, co);
  };
  auto gemmGather = [&](const _Float16* pts, int pc, const float* crd, const float* nx,
                        int Npts, int np, int K, int lidx, int rows, _Float16* oh) {
    int cout = L[lidx].cout, ntN = cout >> 4, cinPad = L[lidx].cinPad;
    int NB = pickNB(ntN, cinPad);
    int blocks = (rows / 128) * (ntN / NB);
    size_t shmem = (size_t)NB * 16 * cinPad * 2;
    if (NB == 4)
      k_gemm_gather_t<4><<<blocks, 256, shmem, stream>>>(pts, pc, crd, nx, gi, Npts, np, K,
          wh[lidx], L[lidx].b, L[lidx].g, L[lidx].be, rows, cinPad, cout, oh);
    else
      k_gemm_gather_t<2><<<blocks, 256, shmem, stream>>>(pts, pc, crd, nx, gi, Npts, np, K,
          wh[lidx], L[lidx].b, L[lidx].g, L[lidx].be, rows, cinPad, cout, oh);
  };

  // ---- SA1 (MSG, 3 branches) ----
  k_zero_f32<<<cdiv(BSZ * NP1 * 320, 256), 256, 0, stream>>>(pool1, BSZ * NP1 * 320);
  {
    const float rad[3] = {0.1f, 0.2f, 0.4f};
    const int   Ks[3]  = {8, 16, 64};
    const int   ofs[3] = {0, 64, 192};
    for (int br = 0; br < 3; ++br) {
      int lb = br * 3, K = Ks[br], rows = BSZ * NP1 * K;
      k_query_ball<<<cdiv(BSZ * NP1, 256), 256, 0, stream>>>(coords, nx1, N1, NP1, K,
                                                             rad[br] * rad[br], gi);
      gemmGather(normh, 3, coords, nx1, N1, NP1, K, lb, rows, act0);
      gemm(act0, lb + 1, rows, L[lb + 1].cin, 0, act1, nullptr, 0, 0);
      gemm(act1, lb + 2, rows, L[lb + 2].cin, K, nullptr, pool1, 320, ofs[br]);
    }
  }
  k_f32_to_f16<<<cdiv(BSZ * NP1 * 320, 256), 256, 0, stream>>>(pool1, pts2, BSZ * NP1 * 320);

  // ---- SA2 (MSG, 3 branches) ----
  k_zero_f32<<<cdiv(BSZ * NP2 * 640, 256), 256, 0, stream>>>(pool2, BSZ * NP2 * 640);
  {
    const float rad[3] = {0.2f, 0.4f, 0.8f};
    const int   Ks[3]  = {16, 32, 128};
    const int   ofs[3] = {0, 128, 384};
    for (int br = 0; br < 3; ++br) {
      int lb = 9 + br * 3, K = Ks[br], rows = BSZ * NP2 * K;
      k_query_ball<<<cdiv(BSZ * NP2, 256), 256, 0, stream>>>(nx1, nx2, NP1, NP2, K,
                                                             rad[br] * rad[br], gi);
      gemmGather(pts2, 320, nx1, nx2, NP1, NP2, K, lb, rows, act0);
      gemm(act0, lb + 1, rows, L[lb + 1].cin, 0, act1, nullptr, 0, 0);
      gemm(act1, lb + 2, rows, L[lb + 2].cin, K, nullptr, pool2, 640, ofs[br]);
    }
  }

  // ---- SA3 (global) + l3 output ----
  float* l3 = (float*)d_out + 512;          // output layout: [logits 32x16 | l3 32x1024]
  k_zero_f32<<<cdiv(BSZ * 1024, 256), 256, 0, stream>>>(l3, BSZ * 1024);
  k_build_sa3in<<<cdiv(BSZ * NP2 * 672, 256), 256, 0, stream>>>(nx2, pool2, sa3in);
  {
    int rows = BSZ * NP2;                   // 2048
    gemm(sa3in, 18, rows, 672, 0, act0, nullptr, 0, 0);
    gemm(act0, 19, rows, 256, 0, act1, nullptr, 0, 0);
    gemm(act1, 20, rows, 512, NP2, nullptr, l3, 1024, 0);
  }

  // ---- FC head + log_softmax ----
  k_fc_head<<<BSZ, 256, 0, stream>>>(l3, fc1w, fc1b, bn1g, bn1b,
                                     fc2w, fc2b, bn2g, bn2b, fc3w, fc3b, (float*)d_out);
}
